// Model_2259152797702
// MI455X (gfx1250) — matrix operand order repaired and checked by reading, not compiled, not run on hardware
//
#include <hip/hip_runtime.h>

// ---------------- problem constants ----------------
constexpr int LAY = 6;
constexpr int HD  = 1024;   // H
constexpr int AD  = 1024;   // A
constexpr int FD  = 4096;   // I (ffn)
constexpr int BS  = 4;      // B
constexpr int SEQ = 2048;   // T
constexpr int MROWS = BS * SEQ;  // 8192
constexpr float EPSV = 1e-5f;

typedef unsigned short u16;
typedef __attribute__((ext_vector_type(8)))  float  v8f;
typedef __attribute__((ext_vector_type(16))) __bf16 v16bf;

union Frag { uint4 q[2]; v16bf v; };

__device__ __forceinline__ u16 f2bf(float f) {
  unsigned int u = __float_as_uint(f);
  u += 0x7fffu + ((u >> 16) & 1u);     // round-to-nearest-even
  return (u16)(u >> 16);
}
__device__ __forceinline__ unsigned pk2bf(float a, float b) {
  return (unsigned)f2bf(a) | ((unsigned)f2bf(b) << 16);
}

// 64B (4 x b128) async DMA global -> LDS; offset: applies to both sides (ISA 08 §4.4)
__device__ __forceinline__ void async_tile_ld(unsigned lds, const u16* g) {
  unsigned long long va = (unsigned long long)g;
  asm volatile(
      "global_load_async_to_lds_b128 %0, %1, off\n\t"
      "global_load_async_to_lds_b128 %0, %1, off offset:16\n\t"
      "global_load_async_to_lds_b128 %0, %1, off offset:32\n\t"
      "global_load_async_to_lds_b128 %0, %1, off offset:48"
      :: "v"(lds), "v"(va) : "memory");
}
__device__ __forceinline__ void wait_async0() {
  asm volatile("s_wait_asynccnt 0x0" ::: "memory");
}

// ---------------- fp32 -> bf16 weight conversion (once per launch) ----------------
__global__ __launch_bounds__(256) void cvt_bf16_kernel(
    const float* __restrict__ s, u16* __restrict__ d) {
  size_t i = ((size_t)blockIdx.x * 256 + threadIdx.x) * 8;
  float4 a = *(const float4*)(s + i);
  float4 b = *(const float4*)(s + i + 4);
  uint4 o;
  o.x = pk2bf(a.x, a.y); o.y = pk2bf(a.z, a.w);
  o.z = pk2bf(b.x, b.y); o.w = pk2bf(b.z, b.w);
  *(uint4*)(d + i) = o;
}

// ---------------- packed 4-way block reduction (256 threads, wave32) ----------------
__device__ __forceinline__ float4 block_reduce4(float4 v) {
  __shared__ float4 part[8];
  #pragma unroll
  for (int o = 16; o > 0; o >>= 1) {
    v.x += __shfl_xor(v.x, o, 32);
    v.y += __shfl_xor(v.y, o, 32);
    v.z += __shfl_xor(v.z, o, 32);
    v.w += __shfl_xor(v.w, o, 32);
  }
  int lane = threadIdx.x & 31, wave = threadIdx.x >> 5;
  if (lane == 0) part[wave] = v;
  __syncthreads();
  float4 r = make_float4(0.f, 0.f, 0.f, 0.f);
  #pragma unroll
  for (int i = 0; i < 8; ++i) {
    float4 p = part[i];
    r.x += p.x; r.y += p.y; r.z += p.z; r.w += p.w;
  }
  __syncthreads();
  return r;
}

// ---------------- embedding gather + LayerNorm ----------------
__global__ __launch_bounds__(256) void embed_ln_kernel(
    const int* __restrict__ ids, const float* __restrict__ emb,
    const float* __restrict__ w, const float* __restrict__ b,
    float* __restrict__ h) {
  int m = blockIdx.x;
  int col = threadIdx.x * 4;
  const float* src = emb + (size_t)ids[m] * HD;
  float4 x = *(const float4*)(src + col);
  float4 sv = make_float4(x.x + x.y + x.z + x.w,
                          x.x*x.x + x.y*x.y + x.z*x.z + x.w*x.w, 0.f, 0.f);
  float4 s = block_reduce4(sv);
  float mu = s.x / HD;
  float rs = rsqrtf(s.y / HD - mu * mu + EPSV);
  float4 o;
  o.x = (x.x - mu) * rs * w[col+0] + b[col+0];
  o.y = (x.y - mu) * rs * w[col+1] + b[col+1];
  o.z = (x.z - mu) * rs * w[col+2] + b[col+2];
  o.w = (x.w - mu) * rs * w[col+3] + b[col+3];
  *(float4*)(h + (size_t)m * HD + col) = o;
}

// ---------------- final LayerNorm -> d_out ----------------
__global__ __launch_bounds__(256) void final_ln_kernel(
    const float* __restrict__ h, const float* __restrict__ w,
    const float* __restrict__ b, float* __restrict__ out) {
  int m = blockIdx.x;
  int col = threadIdx.x * 4;
  float4 x = *(const float4*)(h + (size_t)m * HD + col);
  float4 sv = make_float4(x.x + x.y + x.z + x.w,
                          x.x*x.x + x.y*x.y + x.z*x.z + x.w*x.w, 0.f, 0.f);
  float4 s = block_reduce4(sv);
  float mu = s.x / HD;
  float rs = rsqrtf(s.y / HD - mu * mu + EPSV);
  float4 o;
  o.x = (x.x - mu) * rs * w[col+0] + b[col+0];
  o.y = (x.y - mu) * rs * w[col+1] + b[col+1];
  o.z = (x.z - mu) * rs * w[col+2] + b[col+2];
  o.w = (x.w - mu) * rs * w[col+3] + b[col+3];
  *(float4*)(out + (size_t)m * HD + col) = o;
}

// ---------------- LN + token-shift mixing -> bf16 GEMM inputs ----------------
__global__ __launch_bounds__(256) void ln_mix_kernel(
    const float* __restrict__ h, const float* __restrict__ lnw, const float* __restrict__ lnb,
    const float* __restrict__ mA, const float* __restrict__ mB, const float* __restrict__ mC,
    u16* __restrict__ xA, u16* __restrict__ xB, u16* __restrict__ xC) {
  int m = blockIdx.x;
  int t = m & (SEQ - 1);
  int col = threadIdx.x * 4;
  float4 cur = *(const float4*)(h + (size_t)m * HD + col);
  float4 prv = make_float4(0.f, 0.f, 0.f, 0.f);
  bool hp = (t != 0);
  if (hp) prv = *(const float4*)(h + (size_t)(m - 1) * HD + col);
  float4 sv;
  sv.x = cur.x + cur.y + cur.z + cur.w;
  sv.y = cur.x*cur.x + cur.y*cur.y + cur.z*cur.z + cur.w*cur.w;
  sv.z = prv.x + prv.y + prv.z + prv.w;
  sv.w = prv.x*prv.x + prv.y*prv.y + prv.z*prv.z + prv.w*prv.w;
  float4 s = block_reduce4(sv);
  float muc = s.x / HD, rsc = rsqrtf(s.y / HD - muc * muc + EPSV);
  float mup = s.z / HD, rsp = rsqrtf(s.w / HD - mup * mup + EPSV);
  float c4[4] = {cur.x, cur.y, cur.z, cur.w};
  float p4[4] = {prv.x, prv.y, prv.z, prv.w};
  #pragma unroll
  for (int j = 0; j < 4; ++j) {
    int cc = col + j;
    float w_ = lnw[cc], b_ = lnb[cc];
    float xc = (c4[j] - muc) * rsc * w_ + b_;
    float xp = hp ? ((p4[j] - mup) * rsp * w_ + b_) : 0.f;
    size_t off = (size_t)m * HD + cc;
    float ma = mA[cc];
    xA[off] = f2bf(ma * xc + (1.f - ma) * xp);
    float mb = mB[cc];
    xB[off] = f2bf(mb * xc + (1.f - mb) * xp);
    if (xC) {
      float mc = mC[cc];
      xC[off] = f2bf(mc * xc + (1.f - mc) * xp);
    }
  }
}

// ---------------- WMMA GEMM: C[M,N] = Abf16[M,K] @ Wbf16[N,K]^T ----------------
// Block tile 128x128, BK=64, 256 threads = 8 waves (4M x 2N).
// Double-buffered LDS fed by GLOBAL_LOAD_ASYNC_TO_LDS_B128 (ASYNCcnt), so the
// main loop holds no staged data in VGPRs: DMA of tile k+1 overlaps WMMA of tile k.
// EPI: 0 store fp32, 1 sigmoid fp32, 2 relu^2 -> bf16, 3 h += c, 4 h += Scale*c
constexpr int BK  = 64;
constexpr int LDT = 72;   // 64 + 8 pad halves -> 144B row stride, 16B aligned

template <int EPI, int ND, int KD>
__global__ __launch_bounds__(256) void gemm_bf16(
    const u16* __restrict__ Amat, const u16* __restrict__ Wmat,
    void* __restrict__ Out, const float* __restrict__ Scale) {
  __shared__ u16 sA[2][128 * LDT];
  __shared__ u16 sB[2][128 * LDT];

  const int tid   = threadIdx.x;
  const int lane  = tid & 31;
  const int wave  = tid >> 5;
  const int waveM = wave & 3;   // 4 waves over M, 32 rows each
  const int waveN = wave >> 2;  // 2 waves over N, 64 cols each
  const int bm = blockIdx.y * 128;
  const int bn = blockIdx.x * 128;
  const int lr = tid >> 1;          // DMA row 0..127
  const int lc = (tid & 1) * 32;    // DMA chunk (32 halves = 64B)

  const u16* gA = Amat + (size_t)(bm + lr) * KD + lc;
  const u16* gW = Wmat + (size_t)(bn + lr) * KD + lc;
  // LDS destinations: generic pointer low 32 bits == LDS byte offset (ISA 10.2)
  unsigned ldsA[2] = { (unsigned)(size_t)&sA[0][lr * LDT + lc],
                       (unsigned)(size_t)&sA[1][lr * LDT + lc] };
  unsigned ldsB[2] = { (unsigned)(size_t)&sB[0][lr * LDT + lc],
                       (unsigned)(size_t)&sB[1][lr * LDT + lc] };

  v8f acc[2][4];
  #pragma unroll
  for (int i = 0; i < 2; ++i)
    #pragma unroll
    for (int j = 0; j < 4; ++j)
      #pragma unroll
      for (int e = 0; e < 8; ++e) acc[i][j][e] = 0.f;

  // prologue: DMA tile 0 into buffer 0
  async_tile_ld(ldsA[0], gA);
  async_tile_ld(ldsB[0], gW);

  constexpr int NT = KD / BK;
  #pragma unroll 1
  for (int kt = 0; kt < NT; ++kt) {
    wait_async0();        // this thread's DMA for tile kt complete
    __syncthreads();      // all threads' DMA complete; prior tile's readers done

    if (kt + 1 < NT) {    // kick DMA for tile kt+1 into the other buffer
      gA += BK;
      gW += BK;
      int nb = (kt + 1) & 1;
      async_tile_ld(ldsA[nb], gA);
      async_tile_ld(ldsB[nb], gW);
    }

    const u16* cA = sA[kt & 1];
    const u16* cB = sB[kt & 1];

    // two 16x16x32 sub-steps over the 64-wide K tile
    #pragma unroll
    for (int ks = 0; ks < 2; ++ks) {
      const int kof = ks * 32;
      // A fragments: 16x32 (lanes 0-15 K={0..7,16..23}, lanes 16-31 K={8..15,24..31})
      const int kba = kof + (lane >> 4) * 8;
      Frag af[2];
      #pragma unroll
      for (int sm = 0; sm < 2; ++sm) {
        int mr = waveM * 32 + sm * 16 + (lane & 15);
        af[sm].q[0] = *(const uint4*)&cA[mr * LDT + kba];
        af[sm].q[1] = *(const uint4*)&cA[mr * LDT + 16 + kba];
      }
      // B fragments streamed one at a time: 32x16 K-major
      const int kbs = kof + (lane >> 4) * 8;
      #pragma unroll
      for (int sn = 0; sn < 4; ++sn) {
        int n = waveN * 64 + sn * 16 + (lane & 15);
        Frag bf;
        bf.q[0] = *(const uint4*)&cB[n * LDT + kbs];
        bf.q[1] = *(const uint4*)&cB[n * LDT + kbs + 16];
        acc[0][sn] = __builtin_amdgcn_wmma_f32_16x16x32_bf16(
            false, af[0].v, false, bf.v, (short)0, acc[0][sn], false, false);
        acc[1][sn] = __builtin_amdgcn_wmma_f32_16x16x32_bf16(
            false, af[1].v, false, bf.v, (short)0, acc[1][sn], false, false);
      }
    }
  }

  // Epilogue: C layout = lanes 0-15 rows 0-7, lanes 16-31 rows 8-15, col = lane&15.
  const int lcc = lane & 15;
  const int hr  = (lane >> 4) * 8;
  #pragma unroll
  for (int sm = 0; sm < 2; ++sm) {
    #pragma unroll
    for (int sn = 0; sn < 4; ++sn) {
      #pragma unroll
      for (int i = 0; i < 8; ++i) {
        int row = bm + waveM * 32 + sm * 16 + hr + i;
        int col = bn + waveN * 64 + sn * 16 + lcc;
        size_t off = (size_t)row * ND + col;
        float c = acc[sm][sn][i];
        if constexpr (EPI == 0) {
          ((float*)Out)[off] = c;
        } else if constexpr (EPI == 1) {
          ((float*)Out)[off] = 1.f / (1.f + __expf(-c));
        } else if constexpr (EPI == 2) {
          float t = fmaxf(c, 0.f);
          ((u16*)Out)[off] = f2bf(t * t);
        } else if constexpr (EPI == 3) {
          ((float*)Out)[off] += c;
        } else {
          ((float*)Out)[off] += Scale[off] * c;
        }
      }
    }
  }
}

// ---------------- WKV recurrence (sequential over T, fused r*y -> bf16) ----------------
__global__ __launch_bounds__(256) void wkv_kernel(
    const float* __restrict__ kq, const float* __restrict__ vq, const float* __restrict__ rq,
    const float* __restrict__ uvec, const float* __restrict__ tdec,
    u16* __restrict__ ry) {
  int ch = blockIdx.x * 256 + threadIdx.x;   // 0 .. BS*AD-1
  int b  = ch >> 10;                          // / AD
  int a  = ch & (AD - 1);
  float uu = uvec[a];
  float ww = -__expf(tdec[a]);
  float aa = 0.f, bb = 0.f, pp = -1e38f;
  size_t idx = (size_t)b * SEQ * AD + a;
  for (int t = 0; t < SEQ; ++t, idx += AD) {
    float kt = kq[idx], vt = vq[idx];
    float uk = uu + kt;
    float q  = fmaxf(pp, uk);
    float e1 = __expf(uk - q);
    float e2 = __expf(pp - q);
    float out = (aa * e2 + e1 * vt) / (bb * e2 + e1);
    float pw = pp + ww;
    float q2 = fmaxf(pw, kt);
    float s1 = __expf(pw - q2);
    float s2 = __expf(kt - q2);
    aa = s1 * aa + s2 * vt;
    bb = s1 * bb + s2;
    pp = q2;
    ry[idx] = f2bf(rq[idx] * out);
  }
}

// ---------------- host orchestration ----------------
extern "C" void kernel_launch(void* const* d_in, const int* in_sizes, int n_in,
                              void* d_out, int out_size, void* d_ws, size_t ws_size,
                              hipStream_t stream) {
  const int*   ids    = (const int*)d_in[0];
  const float* emb    = (const float*)d_in[1];
  const float* pre_w  = (const float*)d_in[2];
  const float* pre_b  = (const float*)d_in[3];
  const float* post_w = (const float*)d_in[4];
  const float* post_b = (const float*)d_in[5];
  const float* ln1w   = (const float*)d_in[6];
  const float* ln1b   = (const float*)d_in[7];
  const float* ln2w   = (const float*)d_in[8];
  const float* ln2b   = (const float*)d_in[9];
  const float* amk    = (const float*)d_in[10];
  const float* amv    = (const float*)d_in[11];
  const float* amr    = (const float*)d_in[12];
  const float* awk    = (const float*)d_in[13];
  const float* awv    = (const float*)d_in[14];
  const float* awr    = (const float*)d_in[15];
  const float* awo    = (const float*)d_in[16];
  const float* tdec   = (const float*)d_in[17];
  const float* tfir   = (const float*)d_in[18];
  const float* fmk    = (const float*)d_in[19];
  const float* fmr    = (const float*)d_in[20];
  const float* fwk    = (const float*)d_in[21];
  const float* fwr    = (const float*)d_in[22];
  const float* fwv    = (const float*)d_in[23];

  char* ws = (char*)d_ws;
  float* h  = (float*)ws;  ws += (size_t)MROWS * HD * 4;   // 32 MB
  u16*   x0 = (u16*)ws;    ws += (size_t)MROWS * HD * 2;   // 16 MB
  u16*   x1 = (u16*)ws;    ws += (size_t)MROWS * HD * 2;   // 16 MB
  u16*   x2 = (u16*)ws;    ws += (size_t)MROWS * HD * 2;   // 16 MB
  float* kb = (float*)ws;  ws += (size_t)MROWS * AD * 4;   // 32 MB
  float* vb = (float*)ws;  ws += (size_t)MROWS * AD * 4;   // 32 MB
  float* rb = (float*)ws;  ws += (size_t)MROWS * HD * 4;   // 32 MB
  u16*   big = (u16*)ws;   ws += (size_t)MROWS * FD * 2;   // 64 MB (ry, then kk)

  // bf16 weight mirrors (~156 MB), converted once per launch
  const size_t NAH = (size_t)LAY * AD * HD;   // 6.29M elems (awk/awv/awr/awo/fwr)
  const size_t NIH = (size_t)LAY * FD * HD;   // 25.2M elems (fwk/fwv)
  u16* awk_b = (u16*)ws;  ws += NAH * 2;
  u16* awv_b = (u16*)ws;  ws += NAH * 2;
  u16* awr_b = (u16*)ws;  ws += NAH * 2;
  u16* awo_b = (u16*)ws;  ws += NAH * 2;
  u16* fwk_b = (u16*)ws;  ws += NIH * 2;
  u16* fwr_b = (u16*)ws;  ws += NAH * 2;
  u16* fwv_b = (u16*)ws;  ws += NIH * 2;

  dim3 blk(256);
  dim3 gRows(MROWS);
  dim3 gA(AD / 128, MROWS / 128);
  dim3 gH(HD / 128, MROWS / 128);
  dim3 gI(FD / 128, MROWS / 128);

  // weight conversion (bandwidth-bound, ~470 MB total traffic)
  cvt_bf16_kernel<<<NAH / 2048, blk, 0, stream>>>(awk, awk_b);
  cvt_bf16_kernel<<<NAH / 2048, blk, 0, stream>>>(awv, awv_b);
  cvt_bf16_kernel<<<NAH / 2048, blk, 0, stream>>>(awr, awr_b);
  cvt_bf16_kernel<<<NAH / 2048, blk, 0, stream>>>(awo, awo_b);
  cvt_bf16_kernel<<<NIH / 2048, blk, 0, stream>>>(fwk, fwk_b);
  cvt_bf16_kernel<<<NAH / 2048, blk, 0, stream>>>(fwr, fwr_b);
  cvt_bf16_kernel<<<NIH / 2048, blk, 0, stream>>>(fwv, fwv_b);

  embed_ln_kernel<<<gRows, blk, 0, stream>>>(ids, emb, pre_w, pre_b, h);

  for (int l = 0; l < LAY; ++l) {
    // ---- attention (time mixing) ----
    ln_mix_kernel<<<gRows, blk, 0, stream>>>(
        h, ln1w + l * HD, ln1b + l * HD,
        amk + l * HD, amv + l * HD, amr + l * HD, x0, x1, x2);
    gemm_bf16<0, AD, HD><<<gA, blk, 0, stream>>>(x0, awk_b + (size_t)l * AD * HD, kb, nullptr);
    gemm_bf16<0, AD, HD><<<gA, blk, 0, stream>>>(x1, awv_b + (size_t)l * AD * HD, vb, nullptr);
    gemm_bf16<1, AD, HD><<<gA, blk, 0, stream>>>(x2, awr_b + (size_t)l * AD * HD, rb, nullptr);
    wkv_kernel<<<(BS * AD) / 256, blk, 0, stream>>>(kb, vb, rb, tfir + l * AD, tdec + l * AD, big);
    gemm_bf16<3, HD, AD><<<gH, blk, 0, stream>>>(big, awo_b + (size_t)l * HD * AD, h, nullptr);

    // ---- feed forward (channel mixing) ----
    ln_mix_kernel<<<gRows, blk, 0, stream>>>(
        h, ln2w + l * HD, ln2b + l * HD,
        fmk + l * HD, fmr + l * HD, nullptr, x0, x1, nullptr);
    gemm_bf16<2, FD, HD><<<gI, blk, 0, stream>>>(x0, fwk_b + (size_t)l * FD * HD, big, nullptr);
    gemm_bf16<1, HD, HD><<<gH, blk, 0, stream>>>(x1, fwr_b + (size_t)l * HD * HD, rb, nullptr);
    gemm_bf16<4, HD, FD><<<gH, blk, 0, stream>>>(big, fwv_b + (size_t)l * HD * FD, h, rb);
  }

  final_ln_kernel<<<gRows, blk, 0, stream>>>(h, post_w, post_b, (float*)d_out);
}
